// Net_5720896438296
// MI455X (gfx1250) — matrix operand order repaired and checked by reading, not compiled, not run on hardware
//
#include <hip/hip_runtime.h>
#include <hip/hip_bf16.h>

typedef __bf16 bf16;
typedef __attribute__((ext_vector_type(16))) __bf16 v16bf;
typedef __attribute__((ext_vector_type(8)))  float  v8f;

#define NN   100000
#define NE   800000
#define NSV  51200
#define ESV  409600
#define GN   1024
#define OUTC 50
#define HID  128
#define H2   256

// ---------------------------------------------------------------------------
// utility / elementwise kernels
// ---------------------------------------------------------------------------
__global__ void k_zero(float* __restrict__ p, size_t n) {
  size_t t = (size_t)blockIdx.x * blockDim.x + threadIdx.x;
  if (t < n) p[t] = 0.f;
}

__global__ void k_fill(float* __restrict__ p, int n, float v) {
  int t = blockIdx.x * blockDim.x + threadIdx.x;
  if (t < n) p[t] = v;
}

// X[i, 0:64] = emb[ids[i]], X[i, 64:64+Ff] = feats[i]
__global__ void k_concat(const int* __restrict__ ids, const float* __restrict__ feats,
                         const float* __restrict__ emb, float* __restrict__ X,
                         int n, int C, int Ff) {
  size_t t = (size_t)blockIdx.x * blockDim.x + threadIdx.x;
  if (t >= (size_t)n * C) return;
  int i = (int)(t / C), c = (int)(t % C);
  X[t] = (c < 64) ? emb[ids[i] * 64 + c] : feats[(size_t)i * Ff + (c - 64)];
}

// H = (1 + eps) * X   (eps read from device scalar)
__global__ void k_hinit(const float* __restrict__ X, float* __restrict__ H,
                        size_t total, const float* __restrict__ epsp) {
  size_t t = (size_t)blockIdx.x * blockDim.x + threadIdx.x;
  if (t >= total) return;
  H[t] = (1.f + epsp[0]) * X[t];
}

// GIN aggregation: H[dst] += X[src] per edge (one block per edge, one thread per feature)
__global__ void k_gin_edge(const float* __restrict__ X, float* __restrict__ H,
                           const int* __restrict__ ei, int E, int C) {
  int e = blockIdx.x;
  int c = threadIdx.x;
  if (c >= C) return;
  int src = ei[e], dst = ei[E + e];
  atomicAdd(&H[(size_t)dst * C + c], X[(size_t)src * C + c]);
}

// column sums / sums-of-squares in f64 (blockDim == C)
__global__ void k_colstats(const float* __restrict__ X, int M, int C, int rows_per_blk,
                           double* __restrict__ s, double* __restrict__ s2) {
  int c = threadIdx.x;
  int r0 = blockIdx.x * rows_per_blk;
  int r1 = r0 + rows_per_blk; if (r1 > M) r1 = M;
  float a = 0.f, b = 0.f;
  for (int r = r0; r < r1; ++r) {
    float v = X[(size_t)r * C + c];
    a += v; b += v * v;
  }
  atomicAdd(&s[c],  (double)a);
  atomicAdd(&s2[c], (double)b);
}

// scale = g * rsqrt(var + eps); shift = b - mean * scale
__global__ void k_bn_fin(const double* __restrict__ s, const double* __restrict__ s2,
                         const float* __restrict__ g, const float* __restrict__ b,
                         int C, float invM, float* __restrict__ scale, float* __restrict__ shift) {
  int c = blockIdx.x * blockDim.x + threadIdx.x;
  if (c >= C) return;
  float m = (float)(s[c] * (double)invM);
  float v = (float)(s2[c] * (double)invM) - m * m;
  float sc = g[c] * rsqrtf(v + 1e-5f);
  scale[c] = sc;
  shift[c] = b[c] - m * sc;
}

// in-place x = relu(scale[c]*x + shift[c])
__global__ void k_bnrelu(float* __restrict__ X, size_t total, int C,
                         const float* __restrict__ scale, const float* __restrict__ shift) {
  size_t t = (size_t)blockIdx.x * blockDim.x + threadIdx.x;
  if (t >= total) return;
  int c = (int)(t % C);
  float v = fmaf(X[t], scale[c], shift[c]);
  X[t] = fmaxf(v, 0.f);
}

// ---------------------------------------------------------------------------
// weight conversion: f32 [K,N] row-major -> bf16 [N,Kpad] (transposed, K zero-padded)
// ---------------------------------------------------------------------------
__global__ void k_wconv(const float* __restrict__ W, int K, int N, int Kpad,
                        bf16* __restrict__ WT) {
  int t = blockIdx.x * blockDim.x + threadIdx.x;
  if (t >= N * Kpad) return;
  int n = t / Kpad, k = t % Kpad;
  WT[(size_t)n * Kpad + k] = (k < K) ? (bf16)W[(size_t)k * N + n] : (bf16)0.f;
}

// ---------------------------------------------------------------------------
// WMMA GEMM: C[M,N] = op(A)[M,K] @ W[K,N] + bias
//   XF (compile time): per-K-column BN scale/shift + relu fused into the
//   f32->bf16 LDS staging. Double-buffered 16x32 A tile: one barrier per
//   K-step, next tile's global loads issued before current tile's WMMAs.
//   Requires K even (pair staging never straddles the K boundary).
// ---------------------------------------------------------------------------
template<int TPW, bool XF>
__global__ __launch_bounds__(256)
void k_gemm_wmma(const float* __restrict__ A, int M, int K, int lda, int Kpad,
                 const bf16* __restrict__ WT, int N,
                 const float* __restrict__ bias,
                 const float* __restrict__ ascale, const float* __restrict__ ashift,
                 float* __restrict__ C) {
  __shared__ bf16 As[2][16][32];
  const int tid  = threadIdx.x;
  const int wave = tid >> 5;
  const int lane = tid & 31;
  const int half = lane >> 4;
  const int m    = lane & 15;
  const int r0   = blockIdx.x * 16;
  // staging: each thread owns one aligned pair (row sr, cols sk, sk+1)
  const int sr = tid >> 4;
  const int sk = (tid & 15) << 1;
  const float* arow = A + (size_t)(r0 + sr) * lda;

  auto stage = [&](int kc, int buf) {
    int gk = kc + sk;
    float v0 = 0.f, v1 = 0.f;
    if (gk < K) {
      const float* ap = arow + gk;
      v0 = ap[0];
      v1 = ap[1];
      __builtin_prefetch(ap + 64, 0, 1);       // global_prefetch_b8, 2 tiles ahead
      if (XF) {
        v0 = fmaxf(fmaf(v0, ascale[gk],     ashift[gk]),     0.f);
        v1 = fmaxf(fmaf(v1, ascale[gk + 1], ashift[gk + 1]), 0.f);
      }
    }
    As[buf][sr][sk]     = (bf16)v0;
    As[buf][sr][sk + 1] = (bf16)v1;
  };

  v8f acc[TPW] = {};
  stage(0, 0);
  for (int kc = 0; kc < Kpad; kc += 32) {
    const int cur = (kc >> 5) & 1;
    __syncthreads();
    if (kc + 32 < Kpad) stage(kc + 32, cur ^ 1);   // overlap next tile with WMMAs

    // A fragment (contiguous K order: k = half*16 + element, matching B)
    v16bf af;
    #pragma unroll
    for (int v = 0; v < 8; ++v) {
      int kb = half * 16 + (v << 1);
      af[2 * v]     = As[cur][m][kb];
      af[2 * v + 1] = As[cur][m][kb + 1];
    }

    #pragma unroll
    for (int t = 0; t < TPW; ++t) {
      int ct = wave + (t << 3);
      int n  = (ct << 4) + m;
      // B fragment: lane's 16 contiguous K values from transposed weights (32B load)
      const v16bf bfg = *reinterpret_cast<const v16bf*>(WT + (size_t)n * Kpad + kc + half * 16);
      acc[t] = __builtin_amdgcn_wmma_f32_16x16x32_bf16(
          false, af, false, bfg, (short)0, acc[t], false, false);
    }
  }

  #pragma unroll
  for (int t = 0; t < TPW; ++t) {
    int ct = wave + (t << 3);
    int n  = (ct << 4) + m;
    float bv = bias ? bias[n] : 0.f;
    #pragma unroll
    for (int v = 0; v < 8; ++v) {
      int row = r0 + v + half * 8;   // C/D layout: VGPR v -> M = v + half*8
      C[(size_t)row * N + n] = acc[t][v] + bv;
    }
  }
}

// ---------------------------------------------------------------------------
// GCN propagation kernels
// ---------------------------------------------------------------------------
__global__ void k_degedge(float* __restrict__ deg, const int* __restrict__ ei,
                          const float* __restrict__ w, int E) {
  int e = blockIdx.x * blockDim.x + threadIdx.x;
  if (e >= E) return;
  atomicAdd(&deg[ei[E + e]], w[e]);
}

__global__ void k_rsqrt_ip(float* __restrict__ p, int n) {
  int t = blockIdx.x * blockDim.x + threadIdx.x;
  if (t >= n) return;
  float d = p[t];
  p[t] = d > 0.f ? rsqrtf(d) : 0.f;
}

// out[i] = dinv[i]^2 * xw[i] + bias  (self-loop + bias)
__global__ void k_propinit(const float* __restrict__ xw, const float* __restrict__ dinv,
                           const float* __restrict__ b, float* __restrict__ out,
                           int n, int C) {
  size_t t = (size_t)blockIdx.x * blockDim.x + threadIdx.x;
  if (t >= (size_t)n * C) return;
  int i = (int)(t / C), c = (int)(t % C);
  float d = dinv[i];
  out[t] = fmaf(d * d, xw[t], b[c]);
}

// out[col] += dinv[row]*w*dinv[col] * xw[row]   (one block per edge, blockDim == C)
__global__ void k_propedge(const float* __restrict__ xw, float* __restrict__ out,
                           const float* __restrict__ dinv, const int* __restrict__ ei,
                           const float* __restrict__ w, int E, int C) {
  int e = blockIdx.x;
  int c = threadIdx.x;
  int row = ei[e], col = ei[E + e];
  float coeff = dinv[row] * w[e] * dinv[col];
  atomicAdd(&out[(size_t)col * C + c], coeff * xw[(size_t)row * C + c]);
}

// ---------------------------------------------------------------------------
// pooling + final
// ---------------------------------------------------------------------------
__global__ void k_poolnode(const float* __restrict__ X, const int* __restrict__ batch,
                           float* __restrict__ xg, float* __restrict__ cnt, int C) {
  int i = blockIdx.x, c = threadIdx.x;
  int g = batch[i];
  atomicAdd(&xg[(size_t)g * C + c], X[(size_t)i * C + c]);
  if (c == 0) atomicAdd(&cnt[g], 1.f);
}

__global__ void k_pooldiv(float* __restrict__ xg, const float* __restrict__ cnt, int C) {
  size_t t = (size_t)blockIdx.x * blockDim.x + threadIdx.x;
  if (t >= (size_t)GN * C) return;
  xg[t] /= fmaxf(cnt[t / C], 1.f);
}

__global__ void k_poolsvc(const float* __restrict__ X, float* __restrict__ xsg, int C) {
  int i = blockIdx.x, c = threadIdx.x;
  atomicAdd(&xsg[(size_t)(i % OUTC) * C + c], X[(size_t)i * C + c]);
}

__global__ void k_final(const float* __restrict__ xg, const float* __restrict__ xsg,
                        float* __restrict__ out, float inv) {
  int t = blockIdx.x * blockDim.x + threadIdx.x;
  if (t >= GN * OUTC) return;
  int g = t / OUTC, o = t % OUTC;
  float s = 0.f;
  #pragma unroll 8
  for (int c = 0; c < HID; ++c) s = fmaf(xg[g * HID + c], xsg[o * HID + c], s);
  s *= inv;
  out[t] = 1.f / (1.f + __expf(-s));
}

// ---------------------------------------------------------------------------
// host orchestration
// ---------------------------------------------------------------------------
static inline int cdiv(size_t a, int b) { return (int)((a + b - 1) / b); }

static void gemm(hipStream_t s, const float* A, int M, int K, int lda, int Kpad,
                 const float* W, bf16* wt, int N, const float* bias,
                 const float* asc, const float* ash, float* C) {
  k_wconv<<<cdiv((size_t)N * Kpad, 256), 256, 0, s>>>(W, K, N, Kpad, wt);
  if (asc) {
    if (N == 256)
      k_gemm_wmma<2, true><<<M / 16, 256, 0, s>>>(A, M, K, lda, Kpad, wt, N, bias, asc, ash, C);
    else
      k_gemm_wmma<1, true><<<M / 16, 256, 0, s>>>(A, M, K, lda, Kpad, wt, N, bias, asc, ash, C);
  } else {
    if (N == 256)
      k_gemm_wmma<2, false><<<M / 16, 256, 0, s>>>(A, M, K, lda, Kpad, wt, N, bias, nullptr, nullptr, C);
    else
      k_gemm_wmma<1, false><<<M / 16, 256, 0, s>>>(A, M, K, lda, Kpad, wt, N, bias, nullptr, nullptr, C);
  }
}

static void bn_stats(hipStream_t s, const float* X, int M, int C,
                     const float* g, const float* b, double* st, float* scale, float* shift) {
  k_zero<<<cdiv(2 * C * 2, 256), 256, 0, s>>>((float*)st, (size_t)2 * C * 2);
  k_colstats<<<cdiv(M, 512), C, 0, s>>>(X, M, C, 512, st, st + C);
  k_bn_fin<<<1, C, 0, s>>>(st, st + C, g, b, C, 1.f / (float)M, scale, shift);
}

extern "C" void kernel_launch(void* const* d_in, const int* in_sizes, int n_in,
                              void* d_out, int out_size, void* d_ws, size_t ws_size,
                              hipStream_t stream) {
  // ---- inputs (setup_inputs dict order) ----
  const int*   node_ids   = (const int*)  d_in[0];
  const float* node_feats = (const float*)d_in[1];
  const int*   edge_index = (const int*)  d_in[2];
  const int*   svc_ids    = (const int*)  d_in[3];
  const float* svc_feats  = (const float*)d_in[4];
  const int*   ei_svc     = (const int*)  d_in[5];
  const float* ew_svc     = (const float*)d_in[6];
  const int*   batch      = (const int*)  d_in[7];
  const float* emb_node   = (const float*)d_in[10];
  const float* emb_svc    = (const float*)d_in[11];
  const float* g0_W1 = (const float*)d_in[12]; const float* g0_b1 = (const float*)d_in[13];
  const float* g0_bng = (const float*)d_in[14]; const float* g0_bnb = (const float*)d_in[15];
  const float* g0_W2 = (const float*)d_in[16]; const float* g0_b2 = (const float*)d_in[17];
  const float* g0_eps = (const float*)d_in[18];
  const float* bn0_g = (const float*)d_in[19]; const float* bn0_b = (const float*)d_in[20];
  const float* g1_W1 = (const float*)d_in[21]; const float* g1_b1 = (const float*)d_in[22];
  const float* g1_bng = (const float*)d_in[23]; const float* g1_bnb = (const float*)d_in[24];
  const float* g1_W2 = (const float*)d_in[25]; const float* g1_b2 = (const float*)d_in[26];
  const float* g1_eps = (const float*)d_in[27];
  const float* bn1_g = (const float*)d_in[28]; const float* bn1_b = (const float*)d_in[29];
  const float* nlin_W = (const float*)d_in[30]; const float* nlin_b = (const float*)d_in[31];
  const float* c0_W = (const float*)d_in[32]; const float* c0_b = (const float*)d_in[33];
  const float* cbn0_g = (const float*)d_in[34]; const float* cbn0_b = (const float*)d_in[35];
  const float* c1_W = (const float*)d_in[36]; const float* c1_b = (const float*)d_in[37];
  const float* cbn1_g = (const float*)d_in[38]; const float* cbn1_b = (const float*)d_in[39];
  const float* sl_W = (const float*)d_in[40]; const float* sl_b = (const float*)d_in[41];
  float* out = (float*)d_out;

  // ---- workspace arena ----
  char* base = (char*)d_ws;
  size_t off = 0;
  auto alloc = [&](size_t bytes) -> char* {
    char* p = base + off;
    off += (bytes + 255) & ~(size_t)255;
    return p;
  };
  const int C0 = 70, CS = 68;
  float* nx0 = (float*)alloc((size_t)NN * C0 * 4);   // node concat input
  float* nh0 = (float*)alloc((size_t)NN * C0 * 4);   // (1+eps)x + agg
  float* nt  = (float*)alloc((size_t)NN * H2 * 4);   // hidden 256
  float* ng0 = (float*)alloc((size_t)NN * HID * 4);  // GIN0 out -> x1 (in place)
  float* nh1 = (float*)alloc((size_t)NN * HID * 4);  // h1, later node_lin out
  float* ng1 = (float*)alloc((size_t)NN * HID * 4);  // GIN1 out -> x2 (in place)
  double* st  = (double*)alloc(512 * 8);             // col sum / sumsq
  float* scsh = (float*)alloc(512 * 4);              // bn scale | shift
  bf16*  wbuf = (bf16*)alloc((size_t)256 * 256 * 2); // transposed bf16 weights
  float* xg  = (float*)alloc((size_t)GN * HID * 4);
  float* cnt = (float*)alloc((size_t)GN * 4);
  float* xsg = (float*)alloc((size_t)OUTC * HID * 4);
  if (off > ws_size) return;  // workspace too small; bail safely

  // service branch aliases the node region {nx0,nh0,nt} (dead by then): 145.2MB < 158.4MB
  char* sb = (char*)nx0; size_t so = 0;
  auto salloc = [&](size_t bytes) -> char* {
    char* p = sb + so; so += (bytes + 255) & ~(size_t)255; return p;
  };
  float* sx0 = (float*)salloc((size_t)NSV * CS * 4);
  float* sxw = (float*)salloc((size_t)NSV * H2 * 4);
  float* spr = (float*)salloc((size_t)NSV * H2 * 4);
  float* sxl = (float*)salloc((size_t)NSV * HID * 4);
  float* sdv = (float*)salloc((size_t)NSV * 4);
  float* sc = scsh, *sh = scsh + 256;

  // ================= node branch (GIN x2) =================
  k_concat<<<cdiv((size_t)NN * C0, 256), 256, 0, stream>>>(node_ids, node_feats, emb_node, nx0, NN, C0, 6);
  // GIN0: h = (1+eps)x + sum_{src->dst} x[src]
  k_hinit<<<cdiv((size_t)NN * C0, 256), 256, 0, stream>>>(nx0, nh0, (size_t)NN * C0, g0_eps);
  k_gin_edge<<<NE, C0, 0, stream>>>(nx0, nh0, edge_index, NE, C0);
  gemm(stream, nh0, NN, C0, C0, 96, g0_W1, wbuf, H2, g0_b1, nullptr, nullptr, nt);
  bn_stats(stream, nt, NN, H2, g0_bng, g0_bnb, st, sc, sh);
  gemm(stream, nt, NN, H2, H2, H2, g0_W2, wbuf, HID, g0_b2, sc, sh, ng0); // BN+relu fused on A
  bn_stats(stream, ng0, NN, HID, bn0_g, bn0_b, st, sc, sh);
  k_bnrelu<<<cdiv((size_t)NN * HID, 256), 256, 0, stream>>>(ng0, (size_t)NN * HID, HID, sc, sh); // -> x1
  // GIN1
  k_hinit<<<cdiv((size_t)NN * HID, 256), 256, 0, stream>>>(ng0, nh1, (size_t)NN * HID, g1_eps);
  k_gin_edge<<<NE, HID, 0, stream>>>(ng0, nh1, edge_index, NE, HID);
  gemm(stream, nh1, NN, HID, HID, HID, g1_W1, wbuf, H2, g1_b1, nullptr, nullptr, nt);
  bn_stats(stream, nt, NN, H2, g1_bng, g1_bnb, st, sc, sh);
  gemm(stream, nt, NN, H2, H2, H2, g1_W2, wbuf, HID, g1_b2, sc, sh, ng1);
  bn_stats(stream, ng1, NN, HID, bn1_g, bn1_b, st, sc, sh);
  k_bnrelu<<<cdiv((size_t)NN * HID, 256), 256, 0, stream>>>(ng1, (size_t)NN * HID, HID, sc, sh); // -> x2
  // node_lin (reuse nh1 as output)
  gemm(stream, ng1, NN, HID, HID, HID, nlin_W, wbuf, HID, nlin_b, nullptr, nullptr, nh1);

  // ================= service branch (GCN x2) =================
  k_concat<<<cdiv((size_t)NSV * CS, 256), 256, 0, stream>>>(svc_ids, svc_feats, emb_svc, sx0, NSV, CS, 4);
  // dinv = rsqrt(1 + sum_in w)
  k_fill<<<cdiv(NSV, 256), 256, 0, stream>>>(sdv, NSV, 1.f);
  k_degedge<<<cdiv(ESV, 256), 256, 0, stream>>>(sdv, ei_svc, ew_svc, ESV);
  k_rsqrt_ip<<<cdiv(NSV, 256), 256, 0, stream>>>(sdv, NSV);
  // GCN0: xw = x @ W ; prop = D^-1/2 A~ D^-1/2 xw + b
  gemm(stream, sx0, NSV, CS, CS, 96, c0_W, wbuf, H2, nullptr, nullptr, nullptr, sxw);
  k_propinit<<<cdiv((size_t)NSV * H2, 256), 256, 0, stream>>>(sxw, sdv, c0_b, spr, NSV, H2);
  k_propedge<<<ESV, H2, 0, stream>>>(sxw, spr, sdv, ei_svc, ew_svc, ESV, H2);
  bn_stats(stream, spr, NSV, H2, cbn0_g, cbn0_b, st, sc, sh);
  k_bnrelu<<<cdiv((size_t)NSV * H2, 256), 256, 0, stream>>>(spr, (size_t)NSV * H2, H2, sc, sh); // -> xs1
  // GCN1
  gemm(stream, spr, NSV, H2, H2, H2, c1_W, wbuf, H2, nullptr, nullptr, nullptr, sxw);
  k_propinit<<<cdiv((size_t)NSV * H2, 256), 256, 0, stream>>>(sxw, sdv, c1_b, spr, NSV, H2);
  k_propedge<<<ESV, H2, 0, stream>>>(sxw, spr, sdv, ei_svc, ew_svc, ESV, H2);
  bn_stats(stream, spr, NSV, H2, cbn1_g, cbn1_b, st, sc, sh);
  k_bnrelu<<<cdiv((size_t)NSV * H2, 256), 256, 0, stream>>>(spr, (size_t)NSV * H2, H2, sc, sh); // -> xs2
  // svc_lin
  gemm(stream, spr, NSV, H2, H2, H2, sl_W, wbuf, HID, sl_b, nullptr, nullptr, sxl);

  // ================= pooling + final =================
  k_zero<<<cdiv((size_t)GN * HID, 256), 256, 0, stream>>>(xg, (size_t)GN * HID);
  k_zero<<<cdiv(GN, 256), 256, 0, stream>>>(cnt, (size_t)GN);
  k_zero<<<cdiv((size_t)OUTC * HID, 256), 256, 0, stream>>>(xsg, (size_t)OUTC * HID);
  k_poolnode<<<NN, HID, 0, stream>>>(nh1, batch, xg, cnt, HID);
  k_pooldiv<<<cdiv((size_t)GN * HID, 256), 256, 0, stream>>>(xg, cnt, HID);
  k_poolsvc<<<NSV, HID, 0, stream>>>(sxl, xsg, HID);
  // xsg holds raw sums; each service slot has exactly NSV/OUTC = 1024 members
  k_final<<<cdiv(GN * OUTC, 256), 256, 0, stream>>>(xg, xsg, out, 1.f / (float)(NSV / OUTC));
}
